// LinearAttention_36361193128118
// MI455X (gfx1250) — matrix operand order repaired and checked by reading, not compiled, not run on hardware
//
#include <hip/hip_runtime.h>
#include <hip/hip_bf16.h>

// ---- CDNA5 WMMA vector types -------------------------------------------------
typedef __attribute__((ext_vector_type(16))) _Float16 v16h;
typedef __attribute__((ext_vector_type(8)))  _Float16 v8h;
typedef __attribute__((ext_vector_type(8)))  float    v8f;

#define B_DIM   16
#define C_DIM   256
#define N_PIX   4096
#define NT      32            // pixels per workgroup tile
#define PAD     8
#define LDSROW  (C_DIM + PAD) // padded K extent of LDS tiles (bank-conflict relief)

__device__ __forceinline__ float elu1(float x) {
    return x >= 0.0f ? x + 1.0f : __expf(x);
}

__device__ __forceinline__ v16h cat8(v8h a, v8h b) {
    return __builtin_shufflevector(a, b, 0,1,2,3,4,5,6,7,8,9,10,11,12,13,14,15);
}

// A fragment (16x32 f16): lane l -> M row; halves 0..7 -> K=hi*8+j, 8..15 -> K=16+hi*8+j
__device__ __forceinline__ v16h load_a_frag(const _Float16* __restrict__ W,
                                            int m, int k0, int hi) {
    const _Float16* ap = W + (size_t)m * C_DIM + k0 + hi * 8;
    v8h a0 = *reinterpret_cast<const v8h*>(ap);
    v8h a1 = *reinterpret_cast<const v8h*>(ap + 16);
    return cat8(a0, a1);
}

// ---- kernel 0: weight f32->f16 + zero reduction accumulators -----------------
__global__ void la_prep_kernel(const float* __restrict__ wq, const float* __restrict__ wo,
                               _Float16* __restrict__ wq16, _Float16* __restrict__ wo16,
                               float* __restrict__ accum /* kv ++ denom, 2*B*C floats */) {
    int i = blockIdx.x * blockDim.x + threadIdx.x;
    if (i < 3 * C_DIM * C_DIM) wq16[i] = (_Float16)wq[i];
    if (i < C_DIM * C_DIM)     wo16[i] = (_Float16)wo[i];
    if (i < 2 * B_DIM * C_DIM) accum[i] = 0.0f;
}

// ---- kernel 1: qkv GEMM (768x32x256) + elu+1 + kv/denom reduction ------------
__global__ __launch_bounds__(256)
void la_qkv_kernel(const float* __restrict__ x, const _Float16* __restrict__ wq,
                   _Float16* __restrict__ qk_ws, float* __restrict__ kv,
                   float* __restrict__ dn) {
    __shared__ __align__(16) _Float16 Xs[NT][LDSROW]; // x tile, [pixel][channel] f16

    const int tid    = threadIdx.x;
    const int nTiles = N_PIX / NT;
    const int b      = blockIdx.x / nTiles;
    const int p0     = (blockIdx.x % nTiles) * NT;

    // stage x[b, :, p0:p0+32] -> LDS (f32 -> f16, transposed to [p][c])
    {
        const float* xb = x + (size_t)b * C_DIM * N_PIX + p0;
        #pragma unroll
        for (int it = 0; it < 8; ++it) {
            int lin = it * 256 + tid;       // 2048 float4s total
            int c   = lin >> 3;
            int p4  = (lin & 7) << 2;
            const float4 v = *reinterpret_cast<const float4*>(xb + (size_t)c * N_PIX + p4);
            Xs[p4 + 0][c] = (_Float16)v.x;
            Xs[p4 + 1][c] = (_Float16)v.y;
            Xs[p4 + 2][c] = (_Float16)v.z;
            Xs[p4 + 3][c] = (_Float16)v.w;
        }
    }
    __syncthreads();

    const int wave = tid >> 5;
    const int lane = tid & 31;
    const int l    = lane & 15;
    const int hi   = lane >> 4;

    // acc[grp][i][nb], grp: 0=q,1=k,2=v.  Wave w owns M tiles {grp*16+2w, grp*16+2w+1}
    // so channel c of k and v share the same lane/slot -> kv reduction in registers,
    // and channel sets are disjoint across waves -> no LDS accumulation needed.
    v8f acc[3][2][2] = {};

    for (int k0 = 0; k0 < C_DIM; k0 += 32) {
        v16h bf[2];
        #pragma unroll
        for (int nb = 0; nb < 2; ++nb) {
            const _Float16* bp = &Xs[nb * 16 + l][k0 + hi * 8];
            bf[nb] = cat8(*reinterpret_cast<const v8h*>(bp),
                          *reinterpret_cast<const v8h*>(bp + 16));
        }
        #pragma unroll
        for (int grp = 0; grp < 3; ++grp) {
            #pragma unroll
            for (int i = 0; i < 2; ++i) {
                const int tile = grp * 16 + 2 * wave + i;
                const int m    = tile * 16 + l;
                v16h af = load_a_frag(wq, m, k0, hi);
                if (k0 + 32 < C_DIM)  // pull next K-slab of weights into WGP cache
                    __builtin_prefetch(wq + (size_t)m * C_DIM + k0 + 32 + hi * 8, 0, 3);
                #pragma unroll
                for (int nb = 0; nb < 2; ++nb) {
                    acc[grp][i][nb] = __builtin_amdgcn_wmma_f32_16x16x32_f16(
                        false, af, false, bf[nb], (short)0, acc[grp][i][nb], false, false);
                }
            }
        }
    }

    // epilogue part 1: q -> elu+1 -> f16 workspace
    #pragma unroll
    for (int i = 0; i < 2; ++i)
        #pragma unroll
        for (int nb = 0; nb < 2; ++nb)
            #pragma unroll
            for (int j = 0; j < 8; ++j) {
                const int c = (2 * wave + i) * 16 + hi * 8 + j;
                const int p = p0 + nb * 16 + l;
                qk_ws[((size_t)b * C_DIM + c) * N_PIX + p] = (_Float16)elu1(acc[0][i][nb][j]);
            }

    // epilogue part 2: kv/denom — combine the two pixel sub-tiles in registers,
    // butterfly-reduce the 16 lanes of each half-wave, one global atomic per channel.
    #pragma unroll
    for (int i = 0; i < 2; ++i) {
        #pragma unroll
        for (int j = 0; j < 8; ++j) {
            const int c = (2 * wave + i) * 16 + hi * 8 + j;
            float skv = 0.0f, sdn = 0.0f;
            #pragma unroll
            for (int nb = 0; nb < 2; ++nb) {
                const float kk = elu1(acc[1][i][nb][j]);
                skv += kk * acc[2][i][nb][j];
                sdn += kk;
            }
            #pragma unroll
            for (int off = 8; off >= 1; off >>= 1) {   // stays within each 16-lane half
                skv += __shfl_xor(skv, off, 16);
                sdn += __shfl_xor(sdn, off, 16);
            }
            if (l == 0) {
                atomicAdd(&kv[b * C_DIM + c], skv);    // global_atomic_add_f32
                atomicAdd(&dn[b * C_DIM + c], sdn);
            }
        }
    }
}

// ---- kernel 2: scale qk, output GEMM (256x32x256), fused LayerNorm -----------
__global__ __launch_bounds__(256)
void la_out_kernel(const _Float16* __restrict__ qk_ws, const _Float16* __restrict__ wo,
                   const float* __restrict__ kv, const float* __restrict__ dn,
                   const float* __restrict__ gamma, const float* __restrict__ beta,
                   float* __restrict__ out) {
    __shared__ __align__(16) _Float16 Bs[NT][LDSROW]; // scaled qk tile, [p][c]
    __shared__ float Ob[C_DIM][NT];                   // GEMM2 result tile
    __shared__ float scale_s[C_DIM];
    __shared__ float mu_s[NT], rs_s[NT];

    const int tid    = threadIdx.x;
    const int nTiles = N_PIX / NT;
    const int b      = blockIdx.x / nTiles;
    const int p0     = (blockIdx.x % nTiles) * NT;

    if (tid < C_DIM)
        scale_s[tid] = kv[b * C_DIM + tid] / fmaxf(dn[b * C_DIM + tid], 1e-6f);
    __syncthreads();

    {   // stage B = qk * scale[c], f16, transposed [p][c]; vector loads along p
        const _Float16* qb = qk_ws + (size_t)b * C_DIM * N_PIX + p0;
        #pragma unroll
        for (int it = 0; it < 4; ++it) {
            int lin = it * 256 + tid;       // 1024 v8h total
            int c   = lin >> 2;
            int p8  = (lin & 3) << 3;
            const v8h v = *reinterpret_cast<const v8h*>(qb + (size_t)c * N_PIX + p8);
            const float s = scale_s[c];
            #pragma unroll
            for (int e = 0; e < 8; ++e)
                Bs[p8 + e][c] = (_Float16)((float)v[e] * s);
        }
    }
    __syncthreads();

    const int wave = tid >> 5;
    const int lane = tid & 31;
    const int l    = lane & 15;
    const int hi   = lane >> 4;

    v8f acc[2][2] = {};
    for (int k0 = 0; k0 < C_DIM; k0 += 32) {
        v16h bf[2];
        #pragma unroll
        for (int nb = 0; nb < 2; ++nb) {
            const _Float16* bp = &Bs[nb * 16 + l][k0 + hi * 8];
            bf[nb] = cat8(*reinterpret_cast<const v8h*>(bp),
                          *reinterpret_cast<const v8h*>(bp + 16));
        }
        #pragma unroll
        for (int i = 0; i < 2; ++i) {
            const int m = (2 * wave + i) * 16 + l;
            v16h af = load_a_frag(wo, m, k0, hi);
            if (k0 + 32 < C_DIM)
                __builtin_prefetch(wo + (size_t)m * C_DIM + k0 + 32 + hi * 8, 0, 3);
            #pragma unroll
            for (int nb = 0; nb < 2; ++nb) {
                acc[i][nb] = __builtin_amdgcn_wmma_f32_16x16x32_f16(
                    false, af, false, bf[nb], (short)0, acc[i][nb], false, false);
            }
        }
    }

    // park the tile in LDS so LayerNorm can see all 256 channels per pixel
    #pragma unroll
    for (int i = 0; i < 2; ++i)
        #pragma unroll
        for (int nb = 0; nb < 2; ++nb)
            #pragma unroll
            for (int j = 0; j < 8; ++j)
                Ob[(2 * wave + i) * 16 + hi * 8 + j][nb * 16 + l] = acc[i][nb][j];
    __syncthreads();

    if (tid < NT) {   // per-pixel mean / rstd over channels
        float s = 0.0f, s2 = 0.0f;
        #pragma unroll 8
        for (int c = 0; c < C_DIM; ++c) {
            float v = Ob[c][tid];
            s += v; s2 += v * v;
        }
        float mu  = s * (1.0f / C_DIM);
        float var = fmaxf(s2 * (1.0f / C_DIM) - mu * mu, 0.0f);
        mu_s[tid] = mu;
        rs_s[tid] = rsqrtf(var + 1e-5f);
    }
    __syncthreads();

    {   // thread t writes channel t, 32 contiguous pixels, float4 stores
        const float g  = gamma[tid];
        const float be = beta[tid];
        float* op = out + ((size_t)b * C_DIM + tid) * N_PIX + p0;
        #pragma unroll
        for (int p = 0; p < NT; p += 4) {
            float4 v;
            v.x = (Ob[tid][p + 0] - mu_s[p + 0]) * rs_s[p + 0] * g + be;
            v.y = (Ob[tid][p + 1] - mu_s[p + 1]) * rs_s[p + 1] * g + be;
            v.z = (Ob[tid][p + 2] - mu_s[p + 2]) * rs_s[p + 2] * g + be;
            v.w = (Ob[tid][p + 3] - mu_s[p + 3]) * rs_s[p + 3] * g + be;
            *reinterpret_cast<float4*>(op + p) = v;
        }
    }
}

// ---- host-side launcher ------------------------------------------------------
extern "C" void kernel_launch(void* const* d_in, const int* in_sizes, int n_in,
                              void* d_out, int out_size, void* d_ws, size_t ws_size,
                              hipStream_t stream) {
    (void)in_sizes; (void)n_in; (void)out_size; (void)ws_size;
    const float* x      = (const float*)d_in[0];
    const float* w_qkv  = (const float*)d_in[1];
    const float* w_out  = (const float*)d_in[2];
    const float* ln_g   = (const float*)d_in[3];
    const float* ln_b   = (const float*)d_in[4];
    float*       out    = (float*)d_out;

    char* ws = (char*)d_ws;
    _Float16* qk_ws = (_Float16*)ws;                       // 16*256*4096 f16 = 32 MiB
    size_t off = (size_t)B_DIM * C_DIM * N_PIX * sizeof(_Float16);
    float* kvp = (float*)(ws + off); off += (size_t)B_DIM * C_DIM * sizeof(float);
    float* dnp = (float*)(ws + off); off += (size_t)B_DIM * C_DIM * sizeof(float);
    _Float16* wq16 = (_Float16*)(ws + off); off += (size_t)3 * C_DIM * C_DIM * sizeof(_Float16);
    _Float16* wo16 = (_Float16*)(ws + off);

    la_prep_kernel<<<(3 * C_DIM * C_DIM + 255) / 256, 256, 0, stream>>>(
        w_qkv, w_out, wq16, wo16, kvp /* kv ++ denom contiguous */);

    dim3 grid(B_DIM * (N_PIX / NT));
    la_qkv_kernel<<<grid, 256, 0, stream>>>(x, wq16, qk_ws, kvp, dnp);
    la_out_kernel<<<grid, 256, 0, stream>>>(qk_ws, wo16, kvp, dnp, ln_g, ln_b, out);
}
